// KalmanNetBlock_53601191854501
// MI455X (gfx1250) — matrix operand order repaired and checked by reading, not compiled, not run on hardware
//
#include <hip/hip_runtime.h>
#include <hip/hip_bf16.h>
#include <cstddef>
#include <cstdint>

// ---------------------------------------------------------------------------
// KalmanNet block for gfx1250 (MI455X).
//  * Dense GEMMs: 64x128 macro-tiles, double-buffered LDS staging via
//    GLOBAL_LOAD_ASYNC_TO_LDS_B128 (ASYNCcnt overlap), consumed by
//    v_wmma_f32_16x16x32_f16 with 2x2 register blocking per wave.
//  * Scan: low-rank algebra -> rank-2 register-resident recurrence.
// ---------------------------------------------------------------------------

#define BB   2
#define SS   1024
#define DD   512
#define HH   8
#define DH   64
#define HID  512
#define FFN  2048
#define MM   (BB * SS)        // 2048 tokens
#define EPSF 1e-6f
#define KC   64               // k-chunk per stage
#define LSTR 72               // padded LDS row stride (halfs): 64 data + 8 pad

typedef __attribute__((ext_vector_type(16))) _Float16 v16h;
typedef __attribute__((ext_vector_type(8)))  _Float16 v8h;
typedef __attribute__((ext_vector_type(8)))  float    v8f;

__device__ __forceinline__ unsigned lds_off_u32(const void* p) {
    // generic pointer to __shared__: low 32 bits are the LDS byte offset
    return (unsigned)(uintptr_t)p;
}

__device__ __forceinline__ float fast_sigmoid(float x) {
    // v_rcp_f32 instead of IEEE division ladder
    return __builtin_amdgcn_rcpf(1.f + __expf(-x));
}

// ---------------------------------------------------------------------------
// Weight convert + transpose:  W[K,N] f32 -> Wt[N,K] f16
// ---------------------------------------------------------------------------
__global__ void convt_kernel(const float* __restrict__ W, _Float16* __restrict__ Wt,
                             int K, int N) {
    int idx = blockIdx.x * blockDim.x + threadIdx.x;
    if (idx >= K * N) return;
    int k = idx / N, n = idx % N;
    Wt[(size_t)n * K + k] = (_Float16)W[idx];
}

// ---------------------------------------------------------------------------
// RMSNorm over rows of length Dcols; optional f32 / f16 outputs.
// ---------------------------------------------------------------------------
template<bool WF, bool WH>
__global__ void rmsnorm_kernel(const float* __restrict__ in,
                               const float* __restrict__ scale,
                               float* __restrict__ outf,
                               _Float16* __restrict__ outh,
                               int Dcols) {
    __shared__ float red[256];
    const int row = blockIdx.x;
    const float* x = in + (size_t)row * Dcols;
    float ss = 0.f;
    for (int i = threadIdx.x; i < Dcols; i += blockDim.x) { float v = x[i]; ss += v * v; }
    red[threadIdx.x] = ss;
    __syncthreads();
    for (int s = blockDim.x >> 1; s > 0; s >>= 1) {
        if ((int)threadIdx.x < s) red[threadIdx.x] += red[threadIdx.x + s];
        __syncthreads();
    }
    const float inv = rsqrtf(red[0] / (float)Dcols + EPSF);
    for (int i = threadIdx.x; i < Dcols; i += blockDim.x) {
        float v = x[i] * inv * scale[i];
        if (WF) outf[(size_t)row * Dcols + i] = v;
        if (WH) outh[(size_t)row * Dcols + i] = (_Float16)v;
    }
}

// ---------------------------------------------------------------------------
// Embedding gather: ae[t,:] = embed[action[t],:]  (f32 + f16 copies)
// ---------------------------------------------------------------------------
__global__ void gather_embed_kernel(const int* __restrict__ action,
                                    const float* __restrict__ embed,
                                    float* __restrict__ aef,
                                    _Float16* __restrict__ aeh) {
    const int t = blockIdx.x;
    const int a = action[t];
    const float* src = embed + (size_t)a * HID;
    for (int i = threadIdx.x; i < HID; i += blockDim.x) {
        float v = src[i];
        aef[(size_t)t * HID + i] = v;
        aeh[(size_t)t * HID + i] = (_Float16)v;
    }
}

// ---------------------------------------------------------------------------
// WMMA GEMM, double-buffered async-LDS staging.
//   C[M,N] = epi( A[M,K](f16) @ Bt[N,K](f16)^T )
// Block = 256 threads = 8 waves -> macro tile 64(M) x 128(N).
// Wave (wm 0..1, wn 0..3) computes 32x32 = 2x2 WMMA tiles.
// Per 64-wide stage: async-copy A slab 64x64 + B slab 128x64 into padded LDS
// (6 x GLOBAL_LOAD_ASYNC_TO_LDS_B128 per thread). Next stage's copies are
// issued before waiting (s_wait_asynccnt 6: per-wave ASYNCcnt is in-order,
// so <=6 outstanding means the previous buffer has landed) -> DMA overlaps
// the 8 WMMAs of the current stage.
// Fragment layouts per CDNA5 ISA 7.12.2:
//  A lane L: row = base + (L&15); elems 0..7 -> k=(L>>4)*8+0..7,
//            elems 8..15 -> k=16+(L>>4)*8+0..7
//  B lane L: col = base + (L&15); elems 0..15 -> same k order as A
//  C vgpr r, lane L: m = base + (L>>4)*8 + r, n = base + (L&15)
// ---------------------------------------------------------------------------
template<int ACT, bool BIAS, bool RES, bool WF, bool WH>
__global__ void gemm_wmma_kernel(const _Float16* __restrict__ A,
                                 const _Float16* __restrict__ Bt,
                                 const float* __restrict__ bias,
                                 const float* __restrict__ res,
                                 float* __restrict__ Cf,
                                 _Float16* __restrict__ Ch,
                                 int M, int N, int K) {
    __shared__ _Float16 lA[2][64 * LSTR];    // 2 x  9.0 KB
    __shared__ _Float16 lB[2][128 * LSTR];   // 2 x 18.0 KB

    const int tid  = threadIdx.x;
    const int lane = tid & 31;
    const int wave = tid >> 5;
    const int wm   = wave >> 2;           // 0..1
    const int wn   = wave & 3;            // 0..3
    const int half = lane >> 4;
    const int l    = lane & 15;

    const int tilesN = N >> 7;
    const int m0 = (blockIdx.x / tilesN) << 6;   // 64-row macro tile
    const int n0 = (blockIdx.x % tilesN) << 7;   // 128-col macro tile

    // staging assignment: A 64 rows x 64 halfs (2 x b128 per thread),
    //                     B 128 rows x 64 halfs (4 x b128 per thread)
    const int ar = tid >> 2;                 // 0..63
    const int aq = (tid & 3) << 4;           // 0,16,32,48 halfs
    const int br = tid >> 1;                 // 0..127
    const int bq = (tid & 1) << 5;           // 0,32 halfs

    const unsigned la0 = lds_off_u32(&lA[0][ar * LSTR + aq]);
    const unsigned la1 = lds_off_u32(&lA[1][ar * LSTR + aq]);
    const unsigned lb0 = lds_off_u32(&lB[0][br * LSTR + bq]);
    const unsigned lb1 = lds_off_u32(&lB[1][br * LSTR + bq]);
    const _Float16* gA = A  + (size_t)(m0 + ar) * K + aq;
    const _Float16* gB = Bt + (size_t)(n0 + br) * K + bq;

    auto stage = [&](unsigned la, unsigned lb, int kk) {
        const _Float16* ga = gA + kk;
        const _Float16* gb = gB + kk;
        asm volatile("global_load_async_to_lds_b128 %0, %1, off" :: "v"(la),      "v"(ga)      : "memory");
        asm volatile("global_load_async_to_lds_b128 %0, %1, off" :: "v"(la + 16), "v"(ga + 8)  : "memory");
        asm volatile("global_load_async_to_lds_b128 %0, %1, off" :: "v"(lb),      "v"(gb)      : "memory");
        asm volatile("global_load_async_to_lds_b128 %0, %1, off" :: "v"(lb + 16), "v"(gb + 8)  : "memory");
        asm volatile("global_load_async_to_lds_b128 %0, %1, off" :: "v"(lb + 32), "v"(gb + 16) : "memory");
        asm volatile("global_load_async_to_lds_b128 %0, %1, off" :: "v"(lb + 48), "v"(gb + 24) : "memory");
    };

    v8f acc[2][2] = {};

    stage(la0, lb0, 0);
    const int nstages = K / KC;
    for (int st = 0; st < nstages; ++st) {
        const int cur = st & 1;
        if (st + 1 < nstages) {
            stage(cur ? la0 : la1, cur ? lb0 : lb1, (st + 1) * KC);
            asm volatile("s_wait_asynccnt 0x6" ::: "memory");   // prev buffer landed
        } else {
            asm volatile("s_wait_asynccnt 0x0" ::: "memory");
        }
        __syncthreads();     // slab visible to all waves

        const _Float16* bufA = &lA[cur][0];
        const _Float16* bufB = &lB[cur][0];
#pragma unroll
        for (int ks = 0; ks < KC; ks += 32) {
            v16h af[2], bf[2];
#pragma unroll
            for (int tm = 0; tm < 2; ++tm) {
                const int row = wm * 32 + tm * 16 + l;
                v8h alo = *(const v8h*)(bufA + row * LSTR + ks + half * 8);
                v8h ahi = *(const v8h*)(bufA + row * LSTR + ks + 16 + half * 8);
#pragma unroll
                for (int i = 0; i < 8; ++i) { af[tm][i] = alo[i]; af[tm][i + 8] = ahi[i]; }
            }
#pragma unroll
            for (int tn = 0; tn < 2; ++tn) {
                const int row = wn * 32 + tn * 16 + l;
                v8h b0 = *(const v8h*)(bufB + row * LSTR + ks + half * 8);
                v8h b1 = *(const v8h*)(bufB + row * LSTR + ks + 16 + half * 8);
#pragma unroll
                for (int i = 0; i < 8; ++i) { bf[tn][i] = b0[i]; bf[tn][i + 8] = b1[i]; }
            }
#pragma unroll
            for (int tm = 0; tm < 2; ++tm)
#pragma unroll
                for (int tn = 0; tn < 2; ++tn)
                    acc[tm][tn] = __builtin_amdgcn_wmma_f32_16x16x32_f16(
                        false, af[tm], false, bf[tn], (short)0, acc[tm][tn], false, false);
        }
        __syncthreads();     // reads done before buffer is re-staged
    }

#pragma unroll
    for (int tm = 0; tm < 2; ++tm) {
#pragma unroll
        for (int tn = 0; tn < 2; ++tn) {
            const int nn = n0 + wn * 32 + tn * 16 + l;
#pragma unroll
            for (int r = 0; r < 8; ++r) {
                const int mm = m0 + wm * 32 + tm * 16 + (half << 3) + r;
                const size_t idx = (size_t)mm * N + nn;
                float v = acc[tm][tn][r];
                if (BIAS)          v += bias[nn];
                if (ACT == 1)      v = v * fast_sigmoid(v);     // silu
                else if (ACT == 2) v = fast_sigmoid(v);         // sigmoid
                if (RES)           v += res[idx];
                if (WF)            Cf[idx] = v;
                if (WH)            Ch[idx] = (_Float16)v;
            }
        }
    }
}

// ---------------------------------------------------------------------------
// Small projection: out[t,j] = sigmoid( in[t,:] . W[:,j] ), N = H = 8
// ---------------------------------------------------------------------------
__global__ void sig_proj_kernel(const float* __restrict__ in,
                                const float* __restrict__ W,
                                float* __restrict__ out, int K, int total) {
    int id = blockIdx.x * blockDim.x + threadIdx.x;
    if (id >= total) return;
    int t = id / HH, j = id % HH;
    const float* row = in + (size_t)t * K;
    float s = 0.f;
    for (int k = 0; k < K; ++k) s += row[k] * W[k * HH + j];
    out[id] = fast_sigmoid(s);
}

// ---------------------------------------------------------------------------
// Per-head L2 normalize: 64-element vectors (grid = tokens*H, block = 64)
// ---------------------------------------------------------------------------
__global__ void headnorm_kernel(const float* __restrict__ in, float* __restrict__ out) {
    __shared__ float red[64];
    const size_t base = (size_t)blockIdx.x * DH;
    float v = in[base + threadIdx.x];
    red[threadIdx.x] = v * v;
    __syncthreads();
    for (int s = 32; s > 0; s >>= 1) {
        if ((int)threadIdx.x < s) red[threadIdx.x] += red[threadIdx.x + s];
        __syncthreads();
    }
    out[base + threadIdx.x] = v / (sqrtf(red[0]) + EPSF);
}

// ---------------------------------------------------------------------------
// Gated-linear recurrence. Low-rank identity:
//   hs_t = am*(hs - b*(hs k)k^T) + bu bv^T + b*(v - (bv.k) bu)k^T,  am=alpha*(1-m)
// One block per (b,h); thread i owns row i of the 64x64 state in registers.
// ---------------------------------------------------------------------------
__global__ void scan_kernel(const float* __restrict__ qn, const float* __restrict__ kn,
                            const float* __restrict__ v,  const float* __restrict__ beta,
                            const float* __restrict__ alpha, const float* __restrict__ bu,
                            const float* __restrict__ bv, const int* __restrict__ mask,
                            const float* __restrict__ carry,
                            float* __restrict__ attn, float* __restrict__ hs_out) {
    __shared__ float sk[DH], sq[DH], sbv[DH];
    const int bh = blockIdx.x;         // b*H + h
    const int b  = bh / HH;
    const int hh = bh % HH;
    const int i  = threadIdx.x;        // row 0..63

    float hsrow[DH];
    const float* cr = carry + ((size_t)bh * DH + i) * DH;
#pragma unroll
    for (int j = 0; j < DH; ++j) hsrow[j] = cr[j];

    for (int s = 0; s < SS; ++s) {
        const size_t t    = (size_t)b * SS + s;
        const size_t base = t * HID + (size_t)hh * DH;
        sk[i]  = kn[base + i];
        sq[i]  = qn[base + i];
        sbv[i] = bv[base + i];
        __syncthreads();

        const float bet = beta[t * HH + hh];
        const float am  = alpha[t * HH + hh] * (1.f - (float)mask[t]);
        const float vi  = v[base + i];
        const float bui = bu[base + i];

        float hsk = 0.f, bvk = 0.f;
#pragma unroll
        for (int j = 0; j < DH; ++j) { hsk += hsrow[j] * sk[j]; bvk += sbv[j] * sk[j]; }
        const float c1 = bet * (vi - bvk * bui) - am * bet * hsk;

        float ai = 0.f;
#pragma unroll
        for (int j = 0; j < DH; ++j) {
            float hv = am * hsrow[j] + c1 * sk[j] + bui * sbv[j];
            hsrow[j] = hv;
            ai += hv * sq[j];
        }
        attn[base + i] = ai;
        __syncthreads();
    }

    float* ho = hs_out + ((size_t)bh * DH + i) * DH;
#pragma unroll
    for (int j = 0; j < DH; ++j) ho[j] = hsrow[j];
}

// ---------------------------------------------------------------------------
// Elementwise f16 multiply: ffn_in = silu(gate) * val
// ---------------------------------------------------------------------------
__global__ void mul_h_kernel(const _Float16* __restrict__ a, const _Float16* __restrict__ b,
                             _Float16* __restrict__ o, int n) {
    int i = blockIdx.x * blockDim.x + threadIdx.x;
    if (i < n) o[i] = (_Float16)((float)a[i] * (float)b[i]);
}

// ---------------------------------------------------------------------------
// Host-side launch
// ---------------------------------------------------------------------------
static inline int gemm_grid(int M, int N) { return (M >> 6) * (N >> 7); }

extern "C" void kernel_launch(void* const* d_in, const int* in_sizes, int n_in,
                              void* d_out, int out_size, void* d_ws, size_t ws_size,
                              hipStream_t stream) {
    const float* x        = (const float*)d_in[0];
    const int*   action   = (const int*)d_in[1];
    const int*   mask     = (const int*)d_in[2];
    const float* carry    = (const float*)d_in[3];
    const float* n1_scale = (const float*)d_in[4];
    const float* wq       = (const float*)d_in[5];
    const float* wk       = (const float*)d_in[6];
    const float* wv       = (const float*)d_in[7];
    const float* w_beta   = (const float*)d_in[8];
    const float* embed    = (const float*)d_in[9];
    const float* w_alpha  = (const float*)d_in[10];
    const float* w_bu     = (const float*)d_in[11];
    const float* w_bv     = (const float*)d_in[12];
    const float* na_scale = (const float*)d_in[13];
    const float* w_out    = (const float*)d_in[14];
    const float* b_out    = (const float*)d_in[15];
    const float* n2_scale = (const float*)d_in[16];
    const float* w_gate   = (const float*)d_in[17];
    const float* w_val    = (const float*)d_in[18];
    const float* w_ffn    = (const float*)d_in[19];

    float* out_hs = (float*)d_out;                              // [B,H,Dh,Dh]
    float* out_y  = (float*)d_out + (size_t)BB * HH * DH * DH;  // [B,S,D]

    // ---- workspace bump allocator (256B aligned) ----
    char* p = (char*)d_ws;
    auto alloc = [&](size_t bytes) -> void* {
        void* r = (void*)p;
        p += (bytes + 255) & ~(size_t)255;
        return r;
    };
    _Float16* wq_t   = (_Float16*)alloc((size_t)DD * HID * 2);
    _Float16* wk_t   = (_Float16*)alloc((size_t)DD * HID * 2);
    _Float16* wv_t   = (_Float16*)alloc((size_t)DD * HID * 2);
    _Float16* wbu_t  = (_Float16*)alloc((size_t)HID * HID * 2);
    _Float16* wbv_t  = (_Float16*)alloc((size_t)HID * HID * 2);
    _Float16* wout_t = (_Float16*)alloc((size_t)HID * DD * 2);
    _Float16* wgate_t= (_Float16*)alloc((size_t)DD * FFN * 2);
    _Float16* wval_t = (_Float16*)alloc((size_t)DD * FFN * 2);
    _Float16* wffn_t = (_Float16*)alloc((size_t)FFN * DD * 2);

    float*    h_f    = (float*)   alloc((size_t)MM * DD * 4);
    _Float16* h_h    = (_Float16*)alloc((size_t)MM * DD * 2);
    float*    ae_f   = (float*)   alloc((size_t)MM * HID * 4);
    _Float16* ae_h   = (_Float16*)alloc((size_t)MM * HID * 2);
    float*    q_f    = (float*)   alloc((size_t)MM * HID * 4);
    float*    k_f    = (float*)   alloc((size_t)MM * HID * 4);
    float*    v_f    = (float*)   alloc((size_t)MM * HID * 4);
    float*    qn_f   = (float*)   alloc((size_t)MM * HID * 4);
    float*    kn_f   = (float*)   alloc((size_t)MM * HID * 4);
    float*    bu_f   = (float*)   alloc((size_t)MM * HID * 4);
    float*    bv_f   = (float*)   alloc((size_t)MM * HID * 4);
    float*    beta_f = (float*)   alloc((size_t)MM * HH * 4);
    float*    alpha_f= (float*)   alloc((size_t)MM * HH * 4);
    float*    attn_f = (float*)   alloc((size_t)MM * HID * 4);
    _Float16* attn_h = (_Float16*)alloc((size_t)MM * HID * 2);
    float*    skip_f = (float*)   alloc((size_t)MM * DD * 4);
    _Float16* h2_h   = (_Float16*)alloc((size_t)MM * DD * 2);
    _Float16* g_h    = (_Float16*)alloc((size_t)MM * FFN * 2);
    _Float16* u_h    = (_Float16*)alloc((size_t)MM * FFN * 2);
    _Float16* ffn_h  = (_Float16*)alloc((size_t)MM * FFN * 2);
    (void)ws_size; (void)in_sizes; (void)n_in; (void)out_size;

    // ---- 1. convert/transpose weights to f16 [N,K] ----
    auto conv = [&](const float* W, _Float16* Wt, int K, int N) {
        convt_kernel<<<(K * N + 255) / 256, 256, 0, stream>>>(W, Wt, K, N);
    };
    conv(wq,    wq_t,   DD,  HID);
    conv(wk,    wk_t,   DD,  HID);
    conv(wv,    wv_t,   DD,  HID);
    conv(w_bu,  wbu_t,  HID, HID);
    conv(w_bv,  wbv_t,  HID, HID);
    conv(w_out, wout_t, HID, DD);
    conv(w_gate,wgate_t,DD,  FFN);
    conv(w_val, wval_t, DD,  FFN);
    conv(w_ffn, wffn_t, FFN, DD);

    // ---- 2. h = rmsnorm(x) ; ae = embed[action] ----
    rmsnorm_kernel<true, true><<<MM, 256, 0, stream>>>(x, n1_scale, h_f, h_h, DD);
    gather_embed_kernel<<<MM, 256, 0, stream>>>(action, embed, ae_f, ae_h);

    // ---- 3. WMMA projections ----
    gemm_wmma_kernel<1,false,false,true,false><<<gemm_grid(MM,HID), 256, 0, stream>>>(
        h_h, wq_t, nullptr, nullptr, q_f, nullptr, MM, HID, DD);      // q = silu
    gemm_wmma_kernel<1,false,false,true,false><<<gemm_grid(MM,HID), 256, 0, stream>>>(
        h_h, wk_t, nullptr, nullptr, k_f, nullptr, MM, HID, DD);      // k = silu
    gemm_wmma_kernel<0,false,false,true,false><<<gemm_grid(MM,HID), 256, 0, stream>>>(
        h_h, wv_t, nullptr, nullptr, v_f, nullptr, MM, HID, DD);      // v
    gemm_wmma_kernel<0,false,false,true,false><<<gemm_grid(MM,HID), 256, 0, stream>>>(
        ae_h, wbu_t, nullptr, nullptr, bu_f, nullptr, MM, HID, HID);  // b_u
    gemm_wmma_kernel<0,false,false,true,false><<<gemm_grid(MM,HID), 256, 0, stream>>>(
        ae_h, wbv_t, nullptr, nullptr, bv_f, nullptr, MM, HID, HID);  // b_v

    // ---- 4. beta / alpha (tiny N=8 sigmoid projections) ----
    sig_proj_kernel<<<(MM * HH + 255) / 256, 256, 0, stream>>>(h_f,  w_beta,  beta_f,  DD,  MM * HH);
    sig_proj_kernel<<<(MM * HH + 255) / 256, 256, 0, stream>>>(ae_f, w_alpha, alpha_f, HID, MM * HH);

    // ---- 5. per-head L2 normalize q, k ----
    headnorm_kernel<<<MM * HH, 64, 0, stream>>>(q_f, qn_f);
    headnorm_kernel<<<MM * HH, 64, 0, stream>>>(k_f, kn_f);

    // ---- 6. recurrence (writes attn + hs[:, -1] -> d_out seg 0) ----
    scan_kernel<<<BB * HH, 64, 0, stream>>>(qn_f, kn_f, v_f, beta_f, alpha_f,
                                            bu_f, bv_f, mask, carry, attn_f, out_hs);

    // ---- 7. attn norm, out-proj + bias + residual x -> skip ----
    rmsnorm_kernel<false, true><<<MM, 256, 0, stream>>>(attn_f, na_scale, nullptr, attn_h, HID);
    gemm_wmma_kernel<0,true,true,true,false><<<gemm_grid(MM,DD), 256, 0, stream>>>(
        attn_h, wout_t, b_out, x, skip_f, nullptr, MM, DD, HID);

    // ---- 8. SwiGLU FFN + residual -> d_out seg 1 ----
    rmsnorm_kernel<false, true><<<MM, 256, 0, stream>>>(skip_f, n2_scale, nullptr, h2_h, DD);
    gemm_wmma_kernel<1,false,false,false,true><<<gemm_grid(MM,FFN), 256, 0, stream>>>(
        h2_h, wgate_t, nullptr, nullptr, nullptr, g_h, MM, FFN, DD);  // silu(gate)
    gemm_wmma_kernel<0,false,false,false,true><<<gemm_grid(MM,FFN), 256, 0, stream>>>(
        h2_h, wval_t, nullptr, nullptr, nullptr, u_h, MM, FFN, DD);   // val
    mul_h_kernel<<<(MM * FFN + 255) / 256, 256, 0, stream>>>(g_h, u_h, ffn_h, MM * FFN);
    gemm_wmma_kernel<0,false,true,true,false><<<gemm_grid(MM,DD), 256, 0, stream>>>(
        ffn_h, wffn_t, nullptr, skip_f, out_y, nullptr, MM, DD, FFN); // + skip
}
